// softattention_65541200937045
// MI455X (gfx1250) — matrix operand order repaired and checked by reading, not compiled, not run on hardware
//
#include <hip/hip_runtime.h>
#include <hip/hip_bf16.h>

// ---------------------------------------------------------------------------
// MI455X (gfx1250) fused attention:
//   c = x@cw + cb ; q,k,v = split(c) ; s = q@k^T * 1/32 + sw ; p = softmax(s)
//   ctx = p@v ; out = ctx@pw + pb
// bf16 operands + f32 WMMA accumulation (v_wmma_f32_16x16x32_bf16).
// B operands stored transposed [N][K] so each lane loads contiguous 16B
// chunks matching the ISA fragment layouts. K-loops use an explicit ping-pong
// double buffer (2x-unrolled body, fresh defs each half -> no cross-iteration
// register copies / hazard NOPs) with sched_group_barrier pinning so each
// step's 10 loads are issued BEFORE the previous step's 4 WMMAs
// (partial s_wait_loadcnt, ~1 K-step of latency hiding per wave).
// score_weights (dominant HBM stream, 67MB) is prefetched (global_prefetch_b8)
// while the score WMMAs run.
// ---------------------------------------------------------------------------

typedef __bf16 bf16_t;
typedef __attribute__((ext_vector_type(8)))  bf16_t v8bf;
typedef __attribute__((ext_vector_type(16))) bf16_t v16bf;
typedef __attribute__((ext_vector_type(8)))  float  v8f;

union V16U { v16bf v; v8bf h[2]; };

#define S_LEN 2048
#define H_DIM 1024

// Scheduler group pinning (AMDGPU IGroupLP). Masks: 0x008 = MFMA/WMMA,
// 0x020 = VMEM read, 0x100 = DS read.
#if __has_builtin(__builtin_amdgcn_sched_group_barrier)
#define SGB(mask, size, id) __builtin_amdgcn_sched_group_barrier(mask, size, id)
#else
#define SGB(mask, size, id)
#endif

// A fragment (16x32 bf16, MxK): lane l holds row m=l%16; half=l/16.
// VGPR0-3 <- K = half*8 + 0..7 ; VGPR4-7 <- K = 16 + half*8 + 0..7
__device__ __forceinline__ v16bf load_frag_a(const bf16_t* __restrict__ base,
                                             int ld, int kbase) {
  const int l = threadIdx.x & 31;
  const int m = l & 15;
  const int half = l >> 4;
  const bf16_t* p = base + (size_t)m * ld + kbase + half * 8;
  V16U u;
  u.h[0] = *(const v8bf*)(p);        // K offsets half*8 + 0..7
  u.h[1] = *(const v8bf*)(p + 16);   // K offsets 16 + half*8 + 0..7
  return u.v;
}

// B fragment (32x16 bf16, KxN) loaded from B^T stored [N][K] row-major:
// lane l holds col n=l%16; same K order as A: half*8 + 0..7, 16 + half*8 + 0..7.
__device__ __forceinline__ v16bf load_frag_b(const bf16_t* __restrict__ base,
                                             int ld, int kbase) {
  const int l = threadIdx.x & 31;
  const int n = l & 15;
  const int half = l >> 4;
  const bf16_t* p = base + (size_t)n * ld + kbase + half * 8;
  V16U u;
  u.h[0] = *(const v8bf*)(p);        // K offsets half*8 + 0..7
  u.h[1] = *(const v8bf*)(p + 16);   // K offsets 16 + half*8 + 0..7
  return u.v;
}

__device__ __forceinline__ v8f wmma_bf16(v16bf a, v16bf b, v8f c) {
  // 8 args: (neg_a, A, neg_b, B, c_mod, C, reuse_a, reuse_b)
  return __builtin_amdgcn_wmma_f32_16x16x32_bf16(false, a, false, b,
                                                 (short)0, c, false, false);
}

// Fragment set for one K-step of a 16(M) x 64(N) tile: 1 A frag + 4 B frags.
struct FragSet { v16bf a, b0, b1, b2, b3; };

__device__ __forceinline__ FragSet load_set(const bf16_t* __restrict__ arow, int lda,
                                            const bf16_t* __restrict__ bt, int ldb,
                                            int kb) {
  FragSet f;
  f.a  = load_frag_a(arow, lda, kb);
  f.b0 = load_frag_b(bt,                     ldb, kb);
  f.b1 = load_frag_b(bt + (size_t)16 * ldb,  ldb, kb);
  f.b2 = load_frag_b(bt + (size_t)32 * ldb,  ldb, kb);
  f.b3 = load_frag_b(bt + (size_t)48 * ldb,  ldb, kb);
  return f;
}

__device__ __forceinline__ void wmma4(const FragSet& f, v8f& a0, v8f& a1,
                                      v8f& a2, v8f& a3) {
  a0 = wmma_bf16(f.a, f.b0, a0);
  a1 = wmma_bf16(f.a, f.b1, a1);
  a2 = wmma_bf16(f.a, f.b2, a2);
  a3 = wmma_bf16(f.a, f.b3, a3);
}

// --------------------------- prep kernels ----------------------------------
__global__ void __launch_bounds__(256) cvt_kernel(const float* __restrict__ src,
                                                  bf16_t* __restrict__ dst, int n) {
  int i = blockIdx.x * 256 + threadIdx.x;
  if (i < n) dst[i] = (bf16_t)src[i];
}

// src[R][C] f32 -> dst[C][R] bf16
__global__ void __launch_bounds__(256) tr_kernel(const float* __restrict__ src,
                                                 bf16_t* __restrict__ dst,
                                                 int R, int C) {
  int i = blockIdx.x * 256 + threadIdx.x;
  if (i < R * C) {
    int r = i / C, c = i % C;
    dst[(size_t)c * R + r] = (bf16_t)src[i];
  }
}

// --------------------------- kernel 1: QKV GEMM ----------------------------
// c[8192,3072] = xb[8192,1024] @ cw[1024,3072] + cb ; route to q,k,vT (bf16)
__global__ void __launch_bounds__(256) qkv_kernel(
    const bf16_t* __restrict__ xb, const bf16_t* __restrict__ cwT,
    const float* __restrict__ cb,
    bf16_t* __restrict__ qbuf, bf16_t* __restrict__ kbuf,
    bf16_t* __restrict__ vT) {
  const int wid = blockIdx.x * 8 + (threadIdx.x >> 5);
  const int mt = wid / 48;         // 512 M tiles of 16 rows
  const int ng = wid % 48;         // 48 N groups of 64 cols
  const int l = threadIdx.x & 31;
  const int half = l >> 4, ln = l & 15;

  const bf16_t* arow = xb + (size_t)mt * 16 * H_DIM;
  const bf16_t* bt   = cwT + (size_t)(ng * 64) * H_DIM;

  v8f a0 = {}, a1 = {}, a2 = {}, a3 = {};
  FragSet f0 = load_set(arow, H_DIM, bt, H_DIM, 0);
  for (int kk = 0; kk < 30; kk += 2) {
    FragSet f1 = load_set(arow, H_DIM, bt, H_DIM, (kk + 1) * 32);
    wmma4(f0, a0, a1, a2, a3);
    SGB(0x020, 10, 0);   // step kk+1 loads before step kk WMMAs
    SGB(0x008, 4, 0);
    f0 = load_set(arow, H_DIM, bt, H_DIM, (kk + 2) * 32);
    wmma4(f1, a0, a1, a2, a3);
    SGB(0x020, 10, 0);
    SGB(0x008, 4, 0);
  }
  {
    FragSet f1 = load_set(arow, H_DIM, bt, H_DIM, 31 * 32);
    wmma4(f0, a0, a1, a2, a3);
    wmma4(f1, a0, a1, a2, a3);
  }

  v8f acc[4] = { a0, a1, a2, a3 };
#pragma unroll
  for (int j = 0; j < 4; ++j) {
    const int f = ng * 64 + j * 16 + ln;
    const float bias = cb[f];
#pragma unroll
    for (int g = 0; g < 8; ++g) {
      const int m = g + 8 * half;            // C layout: VGPR g -> row g+8*(l/16)
      const int r = mt * 16 + m;             // global row = b*2048 + s
      float v = acc[j][g] + bias;
      if (f < 1024) {
        qbuf[(size_t)r * H_DIM + f] = (bf16_t)v;
      } else if (f < 2048) {
        kbuf[(size_t)r * H_DIM + (f - 1024)] = (bf16_t)v;
      } else {
        const int ai = f - 2048;
        const int bi = r >> 11;
        const int s  = r & 2047;
        vT[((size_t)bi * H_DIM + ai) * S_LEN + s] = (bf16_t)v;  // V transposed
      }
    }
  }
}

// ----------------- kernel 2: fused scores+softmax+context ------------------
// One workgroup (8 waves) per (b, 16-query block). LDS: 16x2048 f32 scores,
// 16x2048 bf16 probs, 512 f32 reduction = 198656 bytes (< 320KB WGP LDS).
__global__ void __launch_bounds__(256) attn_kernel(
    const bf16_t* __restrict__ qbuf, const bf16_t* __restrict__ kbuf,
    const bf16_t* __restrict__ vT, const float* __restrict__ sw,
    bf16_t* __restrict__ ctx) {
  extern __shared__ char smem[];
  float*  s_sc  = (float*)smem;                                     // 128 KB
  bf16_t* s_pr  = (bf16_t*)(smem + 16 * S_LEN * 4);                 //  64 KB
  float*  s_red = (float*)(smem + 16 * S_LEN * 4 + 16 * S_LEN * 2); //   2 KB

  const int r0   = blockIdx.x * 16;   // global query row base (= b*2048 + s0)
  const int bb   = r0 >> 11;
  const int wave = threadIdx.x >> 5;
  const int l    = threadIdx.x & 31;
  const int half = l >> 4, ln = l & 15;
  const float scale = 0.03125f;       // rsqrt(1024)

  // Prefetch this block's score_weights slice (16 rows x 2048 f32 = 128KB):
  // the dominant HBM stream, consumed in the phase-1 epilogue. 16 lines/thread.
  {
    const float* swb = sw + (size_t)r0 * S_LEN;
    for (int i = threadIdx.x; i < 4096; i += 256)
      __builtin_prefetch(swb + (size_t)i * 32, 0, 0);   // 128B granularity
  }

  // ---- Phase 1: scores = Q @ K^T * scale + sw -> LDS (f32) ----
  const bf16_t* qrow = qbuf + (size_t)r0 * H_DIM;
  const int keybase0 = wave * 256;    // each wave owns 256 keys
  for (int grp = 0; grp < 4; ++grp) {
    const bf16_t* bt = kbuf + (size_t)(bb * S_LEN + keybase0 + grp * 64) * H_DIM;
    v8f a0 = {}, a1 = {}, a2 = {}, a3 = {};
    FragSet f0 = load_set(qrow, H_DIM, bt, H_DIM, 0);
    for (int kk = 0; kk < 30; kk += 2) {
      FragSet f1 = load_set(qrow, H_DIM, bt, H_DIM, (kk + 1) * 32);
      wmma4(f0, a0, a1, a2, a3);
      SGB(0x020, 10, 0);
      SGB(0x008, 4, 0);
      f0 = load_set(qrow, H_DIM, bt, H_DIM, (kk + 2) * 32);
      wmma4(f1, a0, a1, a2, a3);
      SGB(0x020, 10, 0);
      SGB(0x008, 4, 0);
    }
    {
      FragSet f1 = load_set(qrow, H_DIM, bt, H_DIM, 31 * 32);
      wmma4(f0, a0, a1, a2, a3);
      wmma4(f1, a0, a1, a2, a3);
    }

    v8f acc[4] = { a0, a1, a2, a3 };
#pragma unroll
    for (int j = 0; j < 4; ++j) {
      const int key = keybase0 + grp * 64 + j * 16 + ln;
#pragma unroll
      for (int g = 0; g < 8; ++g) {
        const int m = g + 8 * half;
        float v = acc[j][g] * scale + sw[(size_t)(r0 + m) * S_LEN + key];
        s_sc[m * S_LEN + key] = v;
      }
    }
  }
  __syncthreads();

  // ---- Phase 2: softmax over 2048 keys (16 threads per row) ----
  {
    const int row = threadIdx.x >> 4;
    const int t16 = threadIdx.x & 15;
    float* sr = s_sc + row * S_LEN + t16 * 128;
    float lmax = -3.4e38f;
    for (int i = 0; i < 128; ++i) lmax = fmaxf(lmax, sr[i]);
    s_red[row * 16 + t16] = lmax;
    __syncthreads();
    float rmax = s_red[row * 16];
    for (int i = 1; i < 16; ++i) rmax = fmaxf(rmax, s_red[row * 16 + i]);
    float lsum = 0.f;
    for (int i = 0; i < 128; ++i) {
      float e = __expf(sr[i] - rmax);
      sr[i] = e;
      lsum += e;
    }
    s_red[256 + row * 16 + t16] = lsum;
    __syncthreads();
    float rsum = 0.f;
    for (int i = 0; i < 16; ++i) rsum += s_red[256 + row * 16 + i];
    const float inv = 1.0f / rsum;
    bf16_t* pr = s_pr + row * S_LEN + t16 * 128;
    for (int i = 0; i < 128; ++i) pr[i] = (bf16_t)(sr[i] * inv);
  }
  __syncthreads();

  // ---- Phase 3: ctx = P @ V (A frags from LDS, V^T frags from global) ----
  const int fbase = wave * 128;       // each wave owns 128 features
  for (int grp = 0; grp < 2; ++grp) {
    const bf16_t* bt = vT + ((size_t)bb * H_DIM + fbase + grp * 64) * S_LEN;
    v8f a0 = {}, a1 = {}, a2 = {}, a3 = {};
    FragSet f0 = load_set(s_pr, S_LEN, bt, S_LEN, 0);
    for (int kk = 0; kk < 62; kk += 2) {
      FragSet f1 = load_set(s_pr, S_LEN, bt, S_LEN, (kk + 1) * 32);
      wmma4(f0, a0, a1, a2, a3);
      SGB(0x100, 2, 0);    // next A frag from LDS
      SGB(0x020, 8, 0);    // next 4 B frags from global
      SGB(0x008, 4, 0);    // this step's WMMAs
      f0 = load_set(s_pr, S_LEN, bt, S_LEN, (kk + 2) * 32);
      wmma4(f1, a0, a1, a2, a3);
      SGB(0x100, 2, 0);
      SGB(0x020, 8, 0);
      SGB(0x008, 4, 0);
    }
    {
      FragSet f1 = load_set(s_pr, S_LEN, bt, S_LEN, 63 * 32);
      wmma4(f0, a0, a1, a2, a3);
      wmma4(f1, a0, a1, a2, a3);
    }

    v8f acc[4] = { a0, a1, a2, a3 };
#pragma unroll
    for (int j = 0; j < 4; ++j) {
      const int fcol = fbase + grp * 64 + j * 16 + ln;
#pragma unroll
      for (int g = 0; g < 8; ++g) {
        const int m = g + 8 * half;
        ctx[(size_t)(r0 + m) * H_DIM + fcol] = (bf16_t)acc[j][g];
      }
    }
  }
}

// --------------------------- kernel 3: out proj ----------------------------
__global__ void __launch_bounds__(256) proj_kernel(
    const bf16_t* __restrict__ ctx, const bf16_t* __restrict__ pwT,
    const float* __restrict__ pb, float* __restrict__ out) {
  const int wid = blockIdx.x * 8 + (threadIdx.x >> 5);
  const int mt = wid / 16;            // 512 M tiles
  const int ng = wid % 16;            // 16 N groups of 64
  const int l = threadIdx.x & 31;
  const int half = l >> 4, ln = l & 15;

  const bf16_t* arow = ctx + (size_t)mt * 16 * H_DIM;
  const bf16_t* bt   = pwT + (size_t)(ng * 64) * H_DIM;

  v8f a0 = {}, a1 = {}, a2 = {}, a3 = {};
  FragSet f0 = load_set(arow, H_DIM, bt, H_DIM, 0);
  for (int kk = 0; kk < 30; kk += 2) {
    FragSet f1 = load_set(arow, H_DIM, bt, H_DIM, (kk + 1) * 32);
    wmma4(f0, a0, a1, a2, a3);
    SGB(0x020, 10, 0);
    SGB(0x008, 4, 0);
    f0 = load_set(arow, H_DIM, bt, H_DIM, (kk + 2) * 32);
    wmma4(f1, a0, a1, a2, a3);
    SGB(0x020, 10, 0);
    SGB(0x008, 4, 0);
  }
  {
    FragSet f1 = load_set(arow, H_DIM, bt, H_DIM, 31 * 32);
    wmma4(f0, a0, a1, a2, a3);
    wmma4(f1, a0, a1, a2, a3);
  }

  v8f acc[4] = { a0, a1, a2, a3 };
#pragma unroll
  for (int j = 0; j < 4; ++j) {
    const int p = ng * 64 + j * 16 + ln;
    const float bias = pb[p];
#pragma unroll
    for (int g = 0; g < 8; ++g) {
      const int m = g + 8 * half;
      out[(size_t)(mt * 16 + m) * H_DIM + p] = acc[j][g] + bias;
    }
  }
}

// --------------------------- launcher --------------------------------------
extern "C" void kernel_launch(void* const* d_in, const int* in_sizes, int n_in,
                              void* d_out, int out_size, void* d_ws, size_t ws_size,
                              hipStream_t stream) {
  (void)in_sizes; (void)n_in; (void)out_size; (void)ws_size;
  const float* x  = (const float*)d_in[0];   // [4,2048,1024]
  const float* sw = (const float*)d_in[1];   // [4,2048,2048]
  const float* cw = (const float*)d_in[2];   // [1024,3072]
  const float* cb = (const float*)d_in[3];   // [3072]
  const float* pw = (const float*)d_in[4];   // [1024,1024]
  const float* pb = (const float*)d_in[5];   // [1024]
  float* out = (float*)d_out;                // [4,2048,1024]

  char* ws = (char*)d_ws;
  const size_t MB = 1024u * 1024u;
  bf16_t* xb  = (bf16_t*)(ws);               // 16 MB  [8192,1024]
  bf16_t* cwT = (bf16_t*)(ws + 16 * MB);     //  6 MB  [3072,1024]
  bf16_t* pwT = (bf16_t*)(ws + 22 * MB);     //  2 MB  [1024,1024]
  bf16_t* qb  = (bf16_t*)(ws + 24 * MB);     // 16 MB  [8192,1024]
  bf16_t* kb  = (bf16_t*)(ws + 40 * MB);     // 16 MB  [8192,1024]
  bf16_t* vT  = (bf16_t*)(ws + 56 * MB);     // 16 MB  [4,1024,2048]
  bf16_t* ctx = (bf16_t*)(ws + 72 * MB);     // 16 MB  [8192,1024]  (total 88 MB)

  // prep: bf16 conversion + weight transposes
  cvt_kernel<<<8192 * 1024 / 256, 256, 0, stream>>>(x, xb, 8192 * 1024);
  tr_kernel<<<1024 * 3072 / 256, 256, 0, stream>>>(cw, cwT, 1024, 3072);
  tr_kernel<<<1024 * 1024 / 256, 256, 0, stream>>>(pw, pwT, 1024, 1024);

  // QKV projection: 512 Mtiles x 48 Ngroups = 24576 wave tasks / 8 per block
  qkv_kernel<<<3072, 256, 0, stream>>>(xb, cwT, cb, qb, kb, vT);

  // fused attention: one block per 16-query row block
  const size_t lds = 16 * S_LEN * 4 + 16 * S_LEN * 2 + 512 * 4;  // 198656 B
  attn_kernel<<<512, 256, lds, stream>>>(qb, kb, vT, sw, ctx);

  // output projection: 512 x 16 = 8192 wave tasks / 8 per block
  proj_kernel<<<1024, 256, 0, stream>>>(ctx, pwT, pb, out);
}
